// TraceAttention_87033217286727
// MI455X (gfx1250) — compile-verified, matrix operand order checked by reading, not run on hardware
//
#include <hip/hip_runtime.h>
#include <hip/hip_bf16.h>
#include <stdint.h>

// Problem constants (from reference setup_inputs: B=2, L=2049, H=8, E=64, n_vars=32, n_tokens=63)
constexpr int Bsz  = 2;
constexpr int Lseq = 2049;
constexpr int Hn   = 8;
constexpr int En   = 64;
constexpr int NVAR = 32;
constexpr int TPB  = 64;     // tokens per variable block = n_tokens + 1
constexpr float SCALE = 0.125f; // 1/sqrt(64)

typedef __attribute__((ext_vector_type(16))) _Float16     v16h;
typedef __attribute__((ext_vector_type(8)))  float        v8f;
typedef __attribute__((ext_vector_type(4)))  unsigned int v4u;
typedef __attribute__((ext_vector_type(8)))  int          v8i;
typedef __attribute__((ext_vector_type(4)))  int          v4i;

// ---------------------------------------------------------------------------
// WMMA helpers (CDNA5 16x16x32 f16 -> f32). Layouts per ISA 7.12.2:
//   A (16x32, 16-bit): lane&15 = M row; half-wave (lane>=16) offsets K by +8;
//     VGPR j holds K=kb+2j,kb+2j+1 ; VGPR 4+j holds K=kb+16+2j,...
//   B (32x16): mirrored — lane&15 = N column, same K striping.
// ---------------------------------------------------------------------------
__device__ inline v8f wmma_f16(v16h a, v16h b, v8f c) {
  return __builtin_amdgcn_wmma_f32_16x16x32_f16(
      /*neg_a=*/false, a, /*neg_b=*/false, b,
      /*c_mod=*/(short)0, c, /*reuse_a=*/false, /*reuse_b=*/false);
}

__device__ inline v16h frag_row(const _Float16* base, int ld, int row0, int col0, int lane) {
  int m  = row0 + (lane & 15);
  int kb = col0 + ((lane >> 4) << 3);
  const _Float16* p = base + m * ld + kb;
  v16h a;
#pragma unroll
  for (int j = 0; j < 4; ++j) {
    a[2 * j]     = p[2 * j];
    a[2 * j + 1] = p[2 * j + 1];
    a[8 + 2 * j]     = p[16 + 2 * j];
    a[8 + 2 * j + 1] = p[16 + 2 * j + 1];
  }
  return a;
}

__device__ inline v16h frag_col(const _Float16* base, int ld, int row0, int col0, int lane) {
  int n  = col0 + (lane & 15);
  int kb = row0 + ((lane >> 4) << 3);
  v16h b;
#pragma unroll
  for (int j = 0; j < 4; ++j) {
    b[2 * j]     = base[(kb + 2 * j) * ld + n];
    b[2 * j + 1] = base[(kb + 2 * j + 1) * ld + n];
    b[8 + 2 * j]     = base[(kb + 16 + 2 * j) * ld + n];
    b[8 + 2 * j + 1] = base[(kb + 16 + 2 * j + 1) * ld + n];
  }
  return b;
}

// ---------------------------------------------------------------------------
// TDM: 1-D contiguous tile load (nBytes must be a multiple of 4).
// D# layout per CDNA5 ISA sec 8.3/8.4:
//   g0: [1:0]=count=1, [63:32]=lds_addr, [120:64]=global_addr, [127:126]=type=2
//   g1: [17:16]=data_size(2 -> 4B), [79:48]=tensor_dim0, [111:80]=tensor_dim1,
//       [127:112]=tile_dim0, [207:160]=tensor_dim0_stride
// This toolchain's builtin takes 6 args (g0, g1, g2, g3, g4, cpol).
// ---------------------------------------------------------------------------
#if __has_builtin(__builtin_amdgcn_tensor_load_to_lds)
#define HAVE_TDM 1
__device__ inline void tdm_load_1d(void* ldsDst, const void* gSrc, unsigned nBytes) {
  unsigned nDw = nBytes >> 2;
  unsigned ldsAddr = (unsigned)(uintptr_t)ldsDst;      // low 32 bits = LDS byte offset
  unsigned long long ga = (unsigned long long)(uintptr_t)gSrc;

  v4u g0;
  g0[0] = 1u;                                          // count=1, user mode
  g0[1] = ldsAddr;
  g0[2] = (unsigned)(ga & 0xFFFFFFFFu);
  g0[3] = (unsigned)((ga >> 32) & 0x01FFFFFFu) | (2u << 30);  // type=2 ("image")

  v8i g1;
  g1[0] = (int)(2u << 16);                             // data_size = 4B, no multicast
  g1[1] = (int)((nDw & 0xFFFFu) << 16);                // tensor_dim0[15:0]
  g1[2] = (int)(((nDw >> 16) & 0xFFFFu) | (1u << 16)); // tensor_dim0[31:16], tensor_dim1=1
  g1[3] = (int)((nDw & 0xFFFFu) << 16);                // tile_dim0 = nDw
  g1[4] = 0;                                           // tile_dim1/2 unused
  g1[5] = (int)nDw;                                    // tensor_dim0_stride[31:0]
  g1[6] = 0;
  g1[7] = 0;

  v4i z4 = {0, 0, 0, 0};
  v8i z8 = {0, 0, 0, 0, 0, 0, 0, 0};
  __builtin_amdgcn_tensor_load_to_lds(g0, g1, z4, z4, z8, 0);
}
#else
#define HAVE_TDM 0
#endif

// ---------------------------------------------------------------------------
// Kernel 1: partial RoPE (first 32 dims) on Q,K + convert Q,K,V to f16 packed
// as (B,H,L,E) so attention tiles are contiguous per (b,h).
// ---------------------------------------------------------------------------
__global__ void rope_pack_kernel(const float* __restrict__ q,
                                 const float* __restrict__ k,
                                 const float* __restrict__ v,
                                 _Float16* __restrict__ qh,
                                 _Float16* __restrict__ kh,
                                 _Float16* __restrict__ vh) {
  size_t tid = (size_t)blockIdx.x * blockDim.x + threadIdx.x;
  size_t total = (size_t)Bsz * Hn * Lseq * (En / 2);
  if (tid >= total) return;

  int pe = (int)(tid & 31);          // pair index 0..31
  size_t t = tid >> 5;
  int l = (int)(t % Lseq); t /= Lseq;
  int h = (int)(t % Hn);
  int b = (int)(t / Hn);

  size_t iin  = (((size_t)b * Lseq + l) * Hn + h) * En + 2 * pe; // (B,L,H,E)
  size_t iout = (((size_t)b * Hn + h) * Lseq + l) * En + 2 * pe; // (B,H,L,E)

  float q0 = q[iin], q1 = q[iin + 1];
  float k0 = k[iin], k1 = k[iin + 1];

  if (pe < 16) {  // rotary region: rot_dim = 32, half = 16
    int sid = (l == 0) ? 0 : ((l - 1) % TPB);
    float theta = __powf(10000.0f, -(float)pe / 16.0f);
    float ang = (float)sid * theta;
    float s, c;
    __sincosf(ang, &s, &c);
    float nq0 = c * q0 - s * q1, nq1 = c * q1 + s * q0;
    float nk0 = c * k0 - s * k1, nk1 = c * k1 + s * k0;
    q0 = nq0; q1 = nq1; k0 = nk0; k1 = nk1;
  }
  qh[iout] = (_Float16)q0; qh[iout + 1] = (_Float16)q1;
  kh[iout] = (_Float16)k0; kh[iout + 1] = (_Float16)k1;
  vh[iout] = (_Float16)v[iin]; vh[iout + 1] = (_Float16)v[iin + 1];
}

// ---------------------------------------------------------------------------
// Kernel 2: block attention. One workgroup (128 threads = 4 waves) per
// (b, h, var). Wave 0 DMAs the Q/K/V tiles into LDS via the TDM; each wave
// then owns a 16-row strip of the 64x64 output tile.
// ---------------------------------------------------------------------------
__global__ void __launch_bounds__(128)
block_attn_kernel(const _Float16* __restrict__ qh,
                  const _Float16* __restrict__ kh,
                  const _Float16* __restrict__ vh,
                  float* __restrict__ out) {
  __shared__ _Float16 sQ[64 * 64];
  __shared__ _Float16 sK[64 * 64];
  __shared__ _Float16 sV[64 * 64];
  __shared__ _Float16 sP[64 * 64];
  __shared__ float    sS[64 * 64];
  __shared__ _Float16 sK0[64];
  __shared__ float    sV0[64];
  __shared__ float    sS0[64];   // score vs key 0, per local row
  __shared__ float    sP0[64];   // exp prob vs key 0
  __shared__ float    sRow[64];  // softmax denominator

  int var = blockIdx.x % NVAR;
  int h   = (blockIdx.x / NVAR) % Hn;
  int b   = blockIdx.x / (NVAR * Hn);
  int tid  = threadIdx.x;
  int lane = tid & 31;
  int w    = tid >> 5;  // wave id 0..3

  size_t bhBase = ((size_t)b * Hn + h) * (size_t)Lseq * En;
  size_t tile   = bhBase + (size_t)(1 + TPB * var) * En;

  // --- stage 0: global -> LDS (each tile is 64*64 halves = 8192B, contiguous)
#if HAVE_TDM
  if (w == 0) {
    tdm_load_1d(sQ, qh + tile, 64 * 64 * sizeof(_Float16));
    tdm_load_1d(sK, kh + tile, 64 * 64 * sizeof(_Float16));
    tdm_load_1d(sV, vh + tile, 64 * 64 * sizeof(_Float16));
    __builtin_amdgcn_s_wait_tensorcnt(0);
  }
#else
  {
    const uint4* gq = (const uint4*)(qh + tile);
    const uint4* gk = (const uint4*)(kh + tile);
    const uint4* gv = (const uint4*)(vh + tile);
    uint4* lq = (uint4*)sQ;
    uint4* lk = (uint4*)sK;
    uint4* lv = (uint4*)sV;
#pragma unroll
    for (int i = tid; i < 64 * 64 / 8; i += 128) {
      lq[i] = gq[i];
      lk[i] = gk[i];
      lv[i] = gv[i];
    }
  }
#endif
  if (tid < 64) {
    sK0[tid] = kh[bhBase + tid];
    sV0[tid] = (float)vh[bhBase + tid];
  }
  __syncthreads();

  // --- stage 1: S = Q·K^T for this wave's 16 rows (8 WMMAs) ---
  {
    v16h aq0 = frag_row(sQ, 64, 16 * w, 0, lane);
    v16h aq1 = frag_row(sQ, 64, 16 * w, 32, lane);
#pragma unroll
    for (int nt = 0; nt < 4; ++nt) {
      v16h bk0 = frag_row(sK, 64, 16 * nt, 0, lane);   // B = K^T columns
      v16h bk1 = frag_row(sK, 64, 16 * nt, 32, lane);
      v8f acc = {};
      acc = wmma_f16(aq0, bk0, acc);
      acc = wmma_f16(aq1, bk1, acc);
      int r0 = 16 * w + ((lane >> 4) << 3);
      int cn = 16 * nt + (lane & 15);
#pragma unroll
      for (int r = 0; r < 8; ++r) sS[(r0 + r) * 64 + cn] = acc[r];
    }
    // score vs global key 0: row = lane&15, halves combined via shfl
    int m = lane & 15, hf = lane >> 4;
    const _Float16* qrow = &sQ[(16 * w + m) * 64 + 32 * hf];
    float s = 0.0f;
#pragma unroll
    for (int e = 0; e < 32; ++e) s += (float)qrow[e] * (float)sK0[32 * hf + e];
    s += __shfl_xor(s, 16, 32);
    if (hf == 0) sS0[16 * w + m] = s;
  }
  __syncthreads();

  // --- stage 2: softmax over 65 keys; 2 threads per row ---
  {
    int row = tid >> 1, hf = tid & 1;
    const float* srow = &sS[row * 64 + 32 * hf];
    float mx = -__builtin_inff();
#pragma unroll
    for (int c2 = 0; c2 < 32; ++c2) mx = fmaxf(mx, srow[c2]);
    if (hf == 0) mx = fmaxf(mx, sS0[row]);
    mx = fmaxf(mx, __shfl_xor(mx, 1, 32));
    float sum = 0.0f;
#pragma unroll
    for (int c2 = 0; c2 < 32; ++c2) {
      float p = __expf(SCALE * (srow[c2] - mx));
      sum += p;
      sP[row * 64 + 32 * hf + c2] = (_Float16)p;
    }
    if (hf == 0) {
      float p0 = __expf(SCALE * (sS0[row] - mx));
      sum += p0;
      sP0[row] = p0;
    }
    sum += __shfl_xor(sum, 1, 32);
    if (hf == 0) sRow[row] = sum;
  }
  __syncthreads();

  // --- stage 3: O = (P·V + p0·v0) / rowsum (8 WMMAs per wave) ---
  {
    v16h ap0 = frag_row(sP, 64, 16 * w, 0, lane);
    v16h ap1 = frag_row(sP, 64, 16 * w, 32, lane);
#pragma unroll
    for (int nt = 0; nt < 4; ++nt) {
      v16h bv0 = frag_col(sV, 64, 0, 16 * nt, lane);
      v16h bv1 = frag_col(sV, 64, 32, 16 * nt, lane);
      v8f acc = {};
      acc = wmma_f16(ap0, bv0, acc);
      acc = wmma_f16(ap1, bv1, acc);
      int r0 = 16 * w + ((lane >> 4) << 3);
      int e  = 16 * nt + (lane & 15);
#pragma unroll
      for (int r = 0; r < 8; ++r) {
        int row = r0 + r;                       // local row 0..63
        float o = (acc[r] + sP0[row] * sV0[e]) / sRow[row];
        size_t pos = (size_t)(1 + TPB * var + row);
        out[(((size_t)b * Lseq + pos) * Hn + h) * En + e] = o;
      }
    }
  }
}

// ---------------------------------------------------------------------------
// Kernel 3: the global query (position 0) attends to all 2049 keys.
// One workgroup (256 threads) per (b,h).
// ---------------------------------------------------------------------------
__global__ void __launch_bounds__(256)
row0_attn_kernel(const _Float16* __restrict__ qh,
                 const _Float16* __restrict__ kh,
                 const _Float16* __restrict__ vh,
                 float* __restrict__ out) {
  __shared__ float sq[64];
  __shared__ float sp[Lseq];
  __shared__ float red[256];
  __shared__ float sacc[4 * 64];

  int h = blockIdx.x % Hn;
  int b = blockIdx.x / Hn;
  int tid = threadIdx.x;
  size_t bhBase = ((size_t)b * Hn + h) * (size_t)Lseq * En;

  if (tid < 64) sq[tid] = (float)qh[bhBase + tid];  // q at pos 0 (rotated)
  __syncthreads();

  // pass 1: scores + local max
  float mx = -__builtin_inff();
  for (int kidx = tid; kidx < Lseq; kidx += 256) {
    const _Float16* kr = kh + bhBase + (size_t)kidx * En;
    if (kidx + 256 < Lseq) __builtin_prefetch(kr + 256 * En, 0, 0);  // global_prefetch_b8
    float s = 0.0f;
#pragma unroll 8
    for (int e = 0; e < 64; ++e) s += sq[e] * (float)kr[e];
    sp[kidx] = s;
    mx = fmaxf(mx, s);
  }
  red[tid] = mx;
  __syncthreads();
  for (int off = 128; off > 0; off >>= 1) {
    if (tid < off) red[tid] = fmaxf(red[tid], red[tid + off]);
    __syncthreads();
  }
  mx = red[0];
  __syncthreads();

  // pass 2: exp + local sum
  float sum = 0.0f;
  for (int kidx = tid; kidx < Lseq; kidx += 256) {
    float p = __expf(SCALE * (sp[kidx] - mx));
    sp[kidx] = p;
    sum += p;
  }
  red[tid] = sum;
  __syncthreads();
  for (int off = 128; off > 0; off >>= 1) {
    if (tid < off) red[tid] += red[tid + off];
    __syncthreads();
  }
  float denom = red[0];
  __syncthreads();

  // pass 3: weighted value sum, e-parallel with 4-way key chunks
  int e  = tid & 63;
  int ch = tid >> 6;
  float acc = 0.0f;
  for (int kidx = ch; kidx < Lseq; kidx += 4)
    acc += sp[kidx] * (float)vh[bhBase + (size_t)kidx * En + e];
  sacc[ch * 64 + e] = acc;
  __syncthreads();
  if (tid < 64) {
    float o = (sacc[tid] + sacc[64 + tid] + sacc[128 + tid] + sacc[192 + tid]) / denom;
    out[(((size_t)b * Lseq + 0) * Hn + h) * En + tid] = o;
  }
}

// ---------------------------------------------------------------------------
extern "C" void kernel_launch(void* const* d_in, const int* in_sizes, int n_in,
                              void* d_out, int out_size, void* d_ws, size_t ws_size,
                              hipStream_t stream) {
  const float* q = (const float*)d_in[0];
  const float* k = (const float*)d_in[1];
  const float* v = (const float*)d_in[2];
  float* out = (float*)d_out;

  const size_t elems = (size_t)Bsz * Hn * Lseq * En;          // 2,098,176
  const size_t bytesPerBuf = ((elems * sizeof(_Float16)) + 255) & ~(size_t)255;
  _Float16* qh = (_Float16*)d_ws;
  _Float16* kh = (_Float16*)((char*)d_ws + bytesPerBuf);
  _Float16* vh = (_Float16*)((char*)d_ws + 2 * bytesPerBuf);

  // Kernel 1: RoPE + f16 pack
  {
    size_t total = (size_t)Bsz * Hn * Lseq * (En / 2);
    int threads = 256;
    int blocks = (int)((total + threads - 1) / threads);
    rope_pack_kernel<<<blocks, threads, 0, stream>>>(q, k, v, qh, kh, vh);
  }

  // Kernel 2: block-sparse fused attention (WMMA + TDM)
  {
    int blocks = Bsz * Hn * NVAR;  // 512
    block_attn_kernel<<<blocks, 128, 0, stream>>>(qh, kh, vh, out);
  }

  // Kernel 3: global token row 0
  {
    int blocks = Bsz * Hn;  // 16
    row0_attn_kernel<<<blocks, 256, 0, stream>>>(qh, kh, vh, out);
  }
}
